// MultiHeadAttention_80126909874477
// MI455X (gfx1250) — matrix operand order repaired and checked by reading, not compiled, not run on hardware
//
#include <hip/hip_runtime.h>
#include <hip/hip_bf16.h>

typedef _Float16 v16h __attribute__((ext_vector_type(16)));
typedef _Float16 v8h  __attribute__((ext_vector_type(8)));
typedef float    v8f  __attribute__((ext_vector_type(8)));

#define B_  4
#define S_  2048
#define D_  1024
#define H_  16
#define HD_ 64
#define M_  (B_ * S_)   // 8192 rows total

// ---------------------------------------------------------------------------
// helpers
// ---------------------------------------------------------------------------
__device__ inline v8f wmma32(v16h a, v16h b, v8f c) {
  // D = A(16x32 f16) * B(32x16 f16) + C(16x16 f32)
  return __builtin_amdgcn_wmma_f32_16x16x32_f16(false, a, false, b,
                                                (short)0, c, false, false);
}

__device__ inline v8f vzero8() {
  v8f z;
#pragma unroll
  for (int i = 0; i < 8; ++i) z[i] = 0.0f;
  return z;
}

// 16 contiguous halves (B-fragment: k = i + 16*group is contiguous)
__device__ inline v16h ld16h(const _Float16* p) {
  v8h lo = *(const v8h*)p;
  v8h hi = *(const v8h*)(p + 8);
  v16h r;
#pragma unroll
  for (int i = 0; i < 8; ++i) { r[i] = lo[i]; r[i + 8] = hi[i]; }
  return r;
}

// two contiguous 8-half runs (A-fragment: callers pass p1 = p0+8, so k = i + 16*group, contiguous)
__device__ inline v16h ld8x2(const _Float16* p0, const _Float16* p1) {
  v8h lo = *(const v8h*)p0;
  v8h hi = *(const v8h*)p1;
  v16h r;
#pragma unroll
  for (int i = 0; i < 8; ++i) { r[i] = lo[i]; r[i + 8] = hi[i]; }
  return r;
}

// ---------------------------------------------------------------------------
// elementwise conversion kernels
// ---------------------------------------------------------------------------
__global__ void cvt_f32_to_f16(const float* __restrict__ src,
                               _Float16* __restrict__ dst, int n) {
  int i = blockIdx.x * blockDim.x + threadIdx.x;
  if (i < n) dst[i] = (_Float16)src[i];
}

// wt[n*D + k] = (f16) w[k*D + n]   (so B-fragments load contiguously over k)
__global__ void cvt_transpose_w(const float* __restrict__ w,
                                _Float16* __restrict__ wt) {
  int i = blockIdx.x * blockDim.x + threadIdx.x;   // i = n*D + k
  int n = i / D_;
  int k = i - n * D_;
  wt[i] = (_Float16)w[k * D_ + n];
}

// ---------------------------------------------------------------------------
// GEMM: [M_ x 1024] (f16, row-major) @ Wt[1024 x 1024] (f16, [n][k])
//   mode 0: store f16 head-split  [bh][s][64]     (Q, K)
//   mode 1: store f16 transposed  [bh][64][s]     (V)
//   mode 2: store f32 merged + bias                (output projection)
// One wave per 32x64 output tile (B fragments reused across two 16-row
// tiles -> 1.5 b128 loads per WMMA); 8 waves / block.
// ---------------------------------------------------------------------------
__global__ __launch_bounds__(256)
void proj_gemm(const _Float16* __restrict__ A,
               const _Float16* __restrict__ Wt,
               _Float16* __restrict__ out16,
               float* __restrict__ out32,
               const float* __restrict__ bias,
               int mode) {
  const int lane = threadIdx.x & 31;
  const int wv   = threadIdx.x >> 5;
  const int w    = blockIdx.x * 8 + wv;           // 4096 wave-tiles
  const int tile_m = (w >> 4) * 32;               // 256 row-tiles of 32
  const int tile_n = (w & 15) * 64;               // 16 col-tiles of 64
  const int n16 = lane & 15;
  const int g   = lane >> 4;

  v8f accL[4], accH[4];
#pragma unroll
  for (int cf = 0; cf < 4; ++cf) { accL[cf] = vzero8(); accH[cf] = vzero8(); }

  const _Float16* arow0 = A + (size_t)(tile_m + n16) * D_;       // rows m..
  const _Float16* arow1 = arow0 + (size_t)16 * D_;               // rows m+16

  for (int kk = 0; kk < D_; kk += 32) {
    v16h af0 = ld8x2(arow0 + kk + 16 * g, arow0 + kk + 8 + 16 * g);
    v16h af1 = ld8x2(arow1 + kk + 16 * g, arow1 + kk + 8 + 16 * g);
    __builtin_prefetch(arow0 + kk + 256, 0, 3);
#pragma unroll
    for (int cf = 0; cf < 4; ++cf) {
      v16h bf = ld16h(Wt + (size_t)(tile_n + cf * 16 + n16) * D_ + kk + 16 * g);
      accL[cf] = wmma32(af0, bf, accL[cf]);
      accH[cf] = wmma32(af1, bf, accH[cf]);
    }
  }

  const int h = tile_n >> 6;                      // tile_n is a multiple of 64
#pragma unroll
  for (int half = 0; half < 2; ++half) {
#pragma unroll
    for (int r = 0; r < 8; ++r) {
      const int row = tile_m + half * 16 + r + 8 * g;  // global token (b*S+s)
      const int b   = row >> 11;                       // / S_
      const int s   = row & (S_ - 1);
      const v8f* acc = half ? accH : accL;
      if (mode == 0) {
        _Float16* op = out16 + ((size_t)(b * H_ + h) * S_ + s) * HD_;
        op[ 0 + n16] = (_Float16)acc[0][r];
        op[16 + n16] = (_Float16)acc[1][r];
        op[32 + n16] = (_Float16)acc[2][r];
        op[48 + n16] = (_Float16)acc[3][r];
      } else if (mode == 1) {
        _Float16* op = out16 + ((size_t)(b * H_ + h) * HD_) * S_;
        op[(size_t)( 0 + n16) * S_ + s] = (_Float16)acc[0][r];
        op[(size_t)(16 + n16) * S_ + s] = (_Float16)acc[1][r];
        op[(size_t)(32 + n16) * S_ + s] = (_Float16)acc[2][r];
        op[(size_t)(48 + n16) * S_ + s] = (_Float16)acc[3][r];
      } else {
        float* op = out32 + (size_t)row * D_ + tile_n;
        op[ 0 + n16] = acc[0][r] + bias[tile_n +  0 + n16];
        op[16 + n16] = acc[1][r] + bias[tile_n + 16 + n16];
        op[32 + n16] = acc[2][r] + bias[tile_n + 32 + n16];
        op[48 + n16] = acc[3][r] + bias[tile_n + 48 + n16];
      }
    }
  }
}

// ---------------------------------------------------------------------------
// Online-softmax update for one 16-row tile against a 32-key block.
// s0/s1: score fragments (cols kbase..+15 / kbase+16..+31), scaled+masked here.
// Produces p0/p1 (probabilities) and cr (rescale factor for O).
// ---------------------------------------------------------------------------
__device__ inline void softmax_update(const v8f& s0, const v8f& s1,
                                      int rowbase, int kbase, int n16, int g,
                                      float* __restrict__ mi,
                                      float* __restrict__ li,
                                      float* __restrict__ p0,
                                      float* __restrict__ p1,
                                      float* __restrict__ cr) {
#pragma unroll
  for (int r = 0; r < 8; ++r) {
    const int row = rowbase + r + 8 * g;
    float v0 = (kbase + n16      <= row) ? s0[r] * 0.125f : -3.0e38f;
    float v1 = (kbase + 16 + n16 <= row) ? s1[r] * 0.125f : -3.0e38f;
    float mx = fmaxf(v0, v1);
#pragma unroll
    for (int off = 8; off >= 1; off >>= 1)
      mx = fmaxf(mx, __shfl_xor(mx, off, 16));
    const float nm = fmaxf(mi[r], mx);
    cr[r] = __expf(mi[r] - nm);
    mi[r] = nm;
    p0[r] = __expf(v0 - nm);
    p1[r] = __expf(v1 - nm);
    float rs = p0[r] + p1[r];
#pragma unroll
    for (int off = 8; off >= 1; off >>= 1)
      rs += __shfl_xor(rs, off, 16);
    li[r] = li[r] * cr[r] + rs;
  }
}

// ---------------------------------------------------------------------------
// Causal flash attention.
//   qh, kh: [bh][s][64] f16       vt: [bh][64][s] f16 (V transposed)
//   oh:     [b][s][1024] f16 (heads merged, feeds output projection)
// 1 wave per 32-row Q tile (K/V fragments reused across both 16-row halves);
// 4 waves / block. 32-key blocks, online softmax in f32.
// ---------------------------------------------------------------------------
__global__ __launch_bounds__(128)
void attn_fwd(const _Float16* __restrict__ qh,
              const _Float16* __restrict__ kh,
              const _Float16* __restrict__ vt,
              _Float16* __restrict__ oh) {
  __shared__ __align__(32) _Float16 ldsP[4][32 * 32];  // P tile per wave

  const int lane  = threadIdx.x & 31;
  const int wv    = threadIdx.x >> 5;
  const int bh    = blockIdx.x;                  // 0..63
  const int qt    = blockIdx.y * 4 + wv;         // 0..63
  const int qbase = qt * 32;
  const int n16   = lane & 15;
  const int g     = lane >> 4;

  const _Float16* qp = qh + (size_t)bh * S_ * HD_;
  const _Float16* kp = kh + (size_t)bh * S_ * HD_;
  const _Float16* vp = vt + (size_t)bh * HD_ * S_;

  // Q A-fragments: two 16-row tiles x two 32-wide k chunks
  const _Float16* qrow0 = qp + (size_t)(qbase + n16) * HD_;
  const _Float16* qrow1 = qrow0 + (size_t)16 * HD_;
  const v16h a0 = ld8x2(qrow0 +      16 * g, qrow0 +  8 + 16 * g);
  const v16h a1 = ld8x2(qrow0 + 32 + 16 * g, qrow0 + 40 + 16 * g);
  const v16h a2 = ld8x2(qrow1 +      16 * g, qrow1 +  8 + 16 * g);
  const v16h a3 = ld8x2(qrow1 + 32 + 16 * g, qrow1 + 40 + 16 * g);

  v8f o[8];                                      // [tile0: 0..3][tile1: 4..7]
#pragma unroll
  for (int i = 0; i < 8; ++i) o[i] = vzero8();
  float mi0[8], li0[8], mi1[8], li1[8];
#pragma unroll
  for (int r = 0; r < 8; ++r) {
    mi0[r] = -3.0e38f; li0[r] = 0.0f;
    mi1[r] = -3.0e38f; li1[r] = 0.0f;
  }

  _Float16* lp = ldsP[wv];
  const int nblk = (qbase >> 5) + 1;             // causal: keys <= qbase+31
  for (int j = 0; j < nblk; ++j) {
    const int kbase = j * 32;

    // ---- S = Q K^T : each K B-fragment feeds both row tiles ----
    const _Float16* krow0 = kp + (size_t)(kbase + n16) * HD_;
    const _Float16* krow1 = krow0 + 16 * HD_;
    v8f s0, s1, t0, t1;
    {
      v16h bk = ld16h(krow0 + 16 * g);                 // keys n, k 0..31
      s0 = wmma32(a0, bk, vzero8());
      t0 = wmma32(a2, bk, vzero8());
    }
    {
      v16h bk = ld16h(krow0 + 32 + 16 * g);            // keys n, k 32..63
      s0 = wmma32(a1, bk, s0);
      t0 = wmma32(a3, bk, t0);
    }
    {
      v16h bk = ld16h(krow1 + 16 * g);                 // keys 16+n, k 0..31
      s1 = wmma32(a0, bk, vzero8());
      t1 = wmma32(a2, bk, vzero8());
    }
    {
      v16h bk = ld16h(krow1 + 32 + 16 * g);            // keys 16+n, k 32..63
      s1 = wmma32(a1, bk, s1);
      t1 = wmma32(a3, bk, t1);
    }

    // ---- online softmax + P to LDS (C-layout -> A-layout transpose) ----
    float p0[8], p1[8], cr[8];
    softmax_update(s0, s1, qbase, kbase, n16, g, mi0, li0, p0, p1, cr);
#pragma unroll
    for (int r = 0; r < 8; ++r) {
      o[0][r] *= cr[r]; o[1][r] *= cr[r]; o[2][r] *= cr[r]; o[3][r] *= cr[r];
      lp[(r + 8 * g) * 32 + n16]      = (_Float16)p0[r];
      lp[(r + 8 * g) * 32 + 16 + n16] = (_Float16)p1[r];
    }
    softmax_update(t0, t1, qbase + 16, kbase, n16, g, mi1, li1, p0, p1, cr);
#pragma unroll
    for (int r = 0; r < 8; ++r) {
      o[4][r] *= cr[r]; o[5][r] *= cr[r]; o[6][r] *= cr[r]; o[7][r] *= cr[r];
      lp[(16 + r + 8 * g) * 32 + n16]      = (_Float16)p0[r];
      lp[(16 + r + 8 * g) * 32 + 16 + n16] = (_Float16)p1[r];
    }
    asm volatile("" ::: "memory");  // keep DS stores before DS loads (LDS in-order per wave)
    const v16h ap0 = ld8x2(lp + n16 * 32 + 16 * g,
                           lp + n16 * 32 + 8 + 16 * g);
    const v16h ap1 = ld8x2(lp + (16 + n16) * 32 + 16 * g,
                           lp + (16 + n16) * 32 + 8 + 16 * g);

    // ---- O += P V : each V B-fragment feeds both row tiles ----
    const _Float16* vb = vp + kbase + 16 * g;
#pragma unroll
    for (int cf = 0; cf < 4; ++cf) {
      v16h bv = ld16h(vb + (size_t)(cf * 16 + n16) * S_);
      o[cf]     = wmma32(ap0, bv, o[cf]);
      o[4 + cf] = wmma32(ap1, bv, o[4 + cf]);
    }
  }

  // ---- normalize and store merged-head f16 output ----
  const int b = bh / H_;
  const int h = bh - b * H_;
#pragma unroll
  for (int half = 0; half < 2; ++half) {
    const float* lli = half ? li1 : li0;
#pragma unroll
    for (int r = 0; r < 8; ++r) {
      const float inv = 1.0f / lli[r];
      const int row = qbase + half * 16 + r + 8 * g;
      _Float16* op = oh + ((size_t)(b * S_ + row)) * D_ + h * HD_;
      op[ 0 + n16] = (_Float16)(o[half * 4 + 0][r] * inv);
      op[16 + n16] = (_Float16)(o[half * 4 + 1][r] * inv);
      op[32 + n16] = (_Float16)(o[half * 4 + 2][r] * inv);
      op[48 + n16] = (_Float16)(o[half * 4 + 3][r] * inv);
    }
  }
}

// ---------------------------------------------------------------------------
// launch
// ---------------------------------------------------------------------------
extern "C" void kernel_launch(void* const* d_in, const int* in_sizes, int n_in,
                              void* d_out, int out_size, void* d_ws, size_t ws_size,
                              hipStream_t stream) {
  const float* x  = (const float*)d_in[0];
  const float* Wq = (const float*)d_in[1];
  const float* Wk = (const float*)d_in[2];
  const float* Wv = (const float*)d_in[3];
  const float* Wo = (const float*)d_in[4];
  const float* bo = (const float*)d_in[5];
  float* out = (float*)d_out;

  char* ws = (char*)d_ws;
  size_t off = 0;
  auto carve = [&](size_t bytes) -> _Float16* {
    _Float16* p = (_Float16*)(ws + off);
    off += (bytes + 255) & ~(size_t)255;
    return p;
  };
  _Float16* xh  = carve((size_t)M_ * D_ * 2);   // 16 MB
  _Float16* WqT = carve((size_t)D_ * D_ * 2);   //  2 MB
  _Float16* WkT = carve((size_t)D_ * D_ * 2);
  _Float16* WvT = carve((size_t)D_ * D_ * 2);
  _Float16* WoT = carve((size_t)D_ * D_ * 2);
  _Float16* qh  = carve((size_t)M_ * D_ * 2);   // head-split Q
  _Float16* kh  = carve((size_t)M_ * D_ * 2);   // head-split K
  _Float16* vt  = carve((size_t)M_ * D_ * 2);   // V transposed per head
  _Float16* oh  = carve((size_t)M_ * D_ * 2);   // merged attention output

  const int nx = M_ * D_;                       // 8,388,608
  cvt_f32_to_f16<<<(nx + 255) / 256, 256, 0, stream>>>(x, xh, nx);
  const int nw = D_ * D_;
  cvt_transpose_w<<<(nw + 255) / 256, 256, 0, stream>>>(Wq, WqT);
  cvt_transpose_w<<<(nw + 255) / 256, 256, 0, stream>>>(Wk, WkT);
  cvt_transpose_w<<<(nw + 255) / 256, 256, 0, stream>>>(Wv, WvT);
  cvt_transpose_w<<<(nw + 255) / 256, 256, 0, stream>>>(Wo, WoT);

  // 4096 wave-tiles / 8 waves per block = 512 blocks
  proj_gemm<<<512, 256, 0, stream>>>(xh, WqT, qh, nullptr, nullptr, 0);
  proj_gemm<<<512, 256, 0, stream>>>(xh, WkT, kh, nullptr, nullptr, 0);
  proj_gemm<<<512, 256, 0, stream>>>(xh, WvT, vt, nullptr, nullptr, 1);

  // 64 (b,h) x 64 q-tiles of 32 rows / 4 waves per block
  attn_fwd<<<dim3(B_ * H_, S_ / 32 / 4), 128, 0, stream>>>(qh, kh, vt, oh);

  proj_gemm<<<512, 256, 0, stream>>>(oh, WoT, nullptr, out, bo, 2);
}
